// SeriesMemoryBank_41077067219100
// MI455X (gfx1250) — matrix operand order repaired and checked by reading, not compiled, not run on hardware
//
#include <hip/hip_runtime.h>
#include <hip/hip_bf16.h>
#include <math.h>

// Problem constants (from reference setup_inputs)
#define B_Q   2048
#define N_MEM 65536
#define DIM   512
#define ROWP  (DIM + 8)                    // LDS row pitch: +16B kills bank conflicts
#define TOPK  8
#define NCHUNKS 16
#define COLS_PER_CHUNK (N_MEM / NCHUNKS)   // 4096
#define WAVES_PER_WG 4
#define ROWS_PER_WG  (WAVES_PER_WG * 16)   // 64
#define MGROUPS (B_Q / ROWS_PER_WG)        // 32

typedef __attribute__((ext_vector_type(8)))  float  v8f;
typedef __attribute__((ext_vector_type(16))) __bf16 v16bf;
typedef __attribute__((ext_vector_type(8)))  __bf16 v8bf;
typedef __attribute__((ext_vector_type(4)))  __bf16 v4bf;

// Fully-unrolled descending-sorted top-8 insertion (registers only after unroll).
__device__ __forceinline__ void topk_insert(float (&ts)[TOPK], int (&ti)[TOPK],
                                            float s, int idx) {
  if (s <= ts[TOPK - 1]) return;
  float vs = s; int vi = idx;
#pragma unroll
  for (int j = 0; j < TOPK; ++j) {
    if (vs > ts[j]) {
      float tf = ts[j]; int tx = ti[j];
      ts[j] = vs; ti[j] = vi;
      vs = tf; vi = tx;
    }
  }
}

// ---------------------------------------------------------------------------
// Kernel 0: one-shot fp32 -> bf16 hi + residual-lo split of the memory bank.
// Streaming pass (256 MB ~ 11 us at 23.3 TB/s) done ONCE instead of 128x
// inside the GEMM hot loop.
// ---------------------------------------------------------------------------
__launch_bounds__(256)
__global__ void convert_bank_kernel(const float* __restrict__ mem,
                                    __bf16* __restrict__ bhi,
                                    __bf16* __restrict__ blo) {
  const size_t i = ((size_t)blockIdx.x * 256 + threadIdx.x) * 4;
  float4 v = *(const float4*)(mem + i);
  v4bf h, l;
  h[0] = (__bf16)v.x; l[0] = (__bf16)(v.x - (float)h[0]);
  h[1] = (__bf16)v.y; l[1] = (__bf16)(v.y - (float)h[1]);
  h[2] = (__bf16)v.z; l[2] = (__bf16)(v.z - (float)h[2]);
  h[3] = (__bf16)v.w; l[3] = (__bf16)(v.w - (float)h[3]);
  *(v4bf*)(bhi + i) = h;
  *(v4bf*)(blo + i) = l;
}

// ---------------------------------------------------------------------------
// Kernel 1: L2-normalize queries (F.normalize eps=1e-12 semantics) and emit
// fp32 (fallback path) plus optional bf16 hi/lo planes (fast path).
// ---------------------------------------------------------------------------
__launch_bounds__(128)
__global__ void normalize_q_kernel(const float* __restrict__ q,
                                   float* __restrict__ qn,
                                   __bf16* __restrict__ qhi,
                                   __bf16* __restrict__ qlo) {
  const int row = blockIdx.x;
  const int t   = threadIdx.x;          // 128 threads, 4 floats each
  const float* qr = q + (size_t)row * DIM;
  float4 v = *(const float4*)(qr + t * 4);
  float ss = v.x * v.x + v.y * v.y + v.z * v.z + v.w * v.w;
#pragma unroll
  for (int off = 16; off > 0; off >>= 1)
    ss += __shfl_down(ss, off, 32);     // wave32 reduction
  __shared__ float red[4];
  if ((t & 31) == 0) red[t >> 5] = ss;
  __syncthreads();
  float tot = red[0] + red[1] + red[2] + red[3];
  float inv = 1.0f / fmaxf(sqrtf(tot), 1e-12f);
  float4 o;
  o.x = v.x * inv; o.y = v.y * inv; o.z = v.z * inv; o.w = v.w * inv;
  const size_t base = (size_t)row * DIM + t * 4;
  *(float4*)(qn + base) = o;
  if (qhi) {
    v4bf h, l;
    h[0] = (__bf16)o.x; l[0] = (__bf16)(o.x - (float)h[0]);
    h[1] = (__bf16)o.y; l[1] = (__bf16)(o.y - (float)h[1]);
    h[2] = (__bf16)o.z; l[2] = (__bf16)(o.z - (float)h[2]);
    h[3] = (__bf16)o.w; l[3] = (__bf16)(o.w - (float)h[3]);
    *(v4bf*)(qhi + base) = h;
    *(v4bf*)(qlo + base) = l;
  }
}

// ---------------------------------------------------------------------------
// Kernel 2: fused bf16x3 WMMA GEMM (sims = Qn * Mem^T) + mask + running top-8.
// 4 wave32s per workgroup, one 16-row M-subtile each, walking the SAME column
// chunk in lockstep (per-tile barrier) so waves 2-4 hit the WGP$ on B loads
// (4x less L2 traffic). Grid = (NCHUNKS, MGROUPS).
// ---------------------------------------------------------------------------
template <bool PRECONV>
__launch_bounds__(WAVES_PER_WG * 32)
__global__ void gemm_topk_kernel(const float* __restrict__ qn,
                                 const __bf16* __restrict__ qhi_g,
                                 const __bf16* __restrict__ qlo_g,
                                 const float* __restrict__ mem,
                                 const __bf16* __restrict__ bhi_g,
                                 const __bf16* __restrict__ blo_g,
                                 float* __restrict__ psims,
                                 int* __restrict__ pidx) {
  // Per-wave A planes, padded row pitch (bank-conflict-free b128 reads).
  __shared__ __align__(16) __bf16 Ahi[WAVES_PER_WG * 16 * ROWP];  // ~66 KB
  __shared__ __align__(16) __bf16 Alo[WAVES_PER_WG * 16 * ROWP];  // ~66 KB
  __shared__ float stile[WAVES_PER_WG][16][16];                   // 4 KB
  __shared__ float msim[WAVES_PER_WG][16][TOPK];
  __shared__ int   midx[WAVES_PER_WG][16][TOPK];

  const int lane  = threadIdx.x & 31;
  const int w     = threadIdx.x >> 5;            // wave id 0..3
  const int chunk = blockIdx.x;
  const int mg    = blockIdx.y;
  const int row0  = (mg * WAVES_PER_WG + w) * 16;
  const int col0  = chunk * COLS_PER_CHUNK;
  const int aw    = w * 16 * ROWP;               // this wave's LDS A base

  // Stage this wave's 16 query rows as bf16 hi + residual lo in LDS.
  if constexpr (PRECONV) {
    for (int idx = lane * 8; idx < 16 * DIM; idx += 32 * 8) {
      const int r = idx / DIM, c = idx % DIM;
      *(v8bf*)&Ahi[aw + r * ROWP + c] = *(const v8bf*)&qhi_g[(size_t)row0 * DIM + idx];
      *(v8bf*)&Alo[aw + r * ROWP + c] = *(const v8bf*)&qlo_g[(size_t)row0 * DIM + idx];
    }
  } else {
    for (int idx = lane; idx < 16 * DIM; idx += 32) {
      const int r = idx / DIM, c = idx % DIM;
      float v = qn[(size_t)row0 * DIM + idx];
      __bf16 h = (__bf16)v;
      Ahi[aw + r * ROWP + c] = h;
      Alo[aw + r * ROWP + c] = (__bf16)(v - (float)h);
    }
  }
  __syncthreads();

  const int myrow = lane & 15;   // A-row (M) for scan; also N column for C store
  const int hsel  = lane >> 4;   // 0 = lanes 0-15, 1 = lanes 16-31
  const int koff  = hsel * 16;   // A-fragment K offset (contiguous K order, same as B)
  const int cbase = hsel * 8;    // column half this lane scans for top-k

  float ts[TOPK]; int ti[TOPK];
#pragma unroll
  for (int j = 0; j < TOPK; ++j) { ts[j] = -INFINITY; ti[j] = 0; }

  // B fragment (32x16 bf16), this code's contiguous K order: lane L / L+16 hold K 0-15 / 16-31 of column N=L
  // -> 16 contiguous K elements per lane. All 4 waves read IDENTICAL B
  // addresses each tile iteration -> served from WGP$ after the first wave.
  const size_t bRowOff = (size_t)(col0 + myrow) * DIM + hsel * 16;
  const float*  bbaseF = mem   + bRowOff;
  const __bf16* bbaseH = bhi_g + bRowOff;
  const __bf16* bbaseL = blo_g + bRowOff;

  for (int ct = 0; ct < COLS_PER_CHUNK / 16; ++ct) {
    const int nb = col0 + ct * 16;
    const size_t toff = (size_t)ct * 16 * DIM;

    if (w == 0 && ct + 1 < COLS_PER_CHUNK / 16) {
      if constexpr (PRECONV)
        __builtin_prefetch((const void*)(bbaseH + toff + (size_t)16 * DIM), 0, 0);
      else
        __builtin_prefetch((const void*)(bbaseF + toff + (size_t)16 * DIM), 0, 0);
    }

    v8f acc = {};

#pragma unroll 4
    for (int kb = 0; kb < DIM; kb += 32) {
      // --- B fragment: bf16 hi/lo, preconverted (fast) or split in-flight ---
      v16bf bh, bl;
      if constexpr (PRECONV) {
        bh = *(const v16bf*)(bbaseH + toff + kb);   // 32 B -> 2x b128
        bl = *(const v16bf*)(bbaseL + toff + kb);
      } else {
        float bv[16];
#pragma unroll
        for (int j = 0; j < 16; j += 4) {
          float4 t = *(const float4*)(bbaseF + toff + kb + j);
          bv[j] = t.x; bv[j + 1] = t.y; bv[j + 2] = t.z; bv[j + 3] = t.w;
        }
#pragma unroll
        for (int j = 0; j < 16; ++j) {
          __bf16 h = (__bf16)bv[j];
          bh[j] = h;
          bl[j] = (__bf16)(bv[j] - (float)h);
        }
      }

      // --- A fragments from LDS (contiguous K order):
      // low lanes: K kb..kb+15; high lanes: K kb+16..kb+31 (matches B).
      const int a0 = aw + myrow * ROWP + kb + koff;
      v8bf ah0 = *(const v8bf*)&Ahi[a0];
      v8bf ah1 = *(const v8bf*)&Ahi[a0 + 8];
      v8bf al0 = *(const v8bf*)&Alo[a0];
      v8bf al1 = *(const v8bf*)&Alo[a0 + 8];
      v16bf ah = __builtin_shufflevector(ah0, ah1, 0,1,2,3,4,5,6,7,8,9,10,11,12,13,14,15);
      v16bf al = __builtin_shufflevector(al0, al1, 0,1,2,3,4,5,6,7,8,9,10,11,12,13,14,15);

      // bf16x3: hi*hi + hi*lo + lo*hi, fp32 accumulation (~2^-16 accurate)
      acc = __builtin_amdgcn_wmma_f32_16x16x32_bf16(false, ah, false, bh,
                                                    (short)0, acc, false, false);
      acc = __builtin_amdgcn_wmma_f32_16x16x32_bf16(false, ah, false, bl,
                                                    (short)0, acc, false, false);
      acc = __builtin_amdgcn_wmma_f32_16x16x32_bf16(false, al, false, bh,
                                                    (short)0, acc, false, false);
    }

    // C/D layout: VGPR v, lanes 0-15 -> M=v, N=lane; lanes 16-31 -> M=v+8.
    // The barrier also keeps the 4 waves in lockstep for WGP$ reuse on B.
    __syncthreads();
#pragma unroll
    for (int v = 0; v < 8; ++v)
      stile[w][v + hsel * 8][myrow] = acc[v];
    __syncthreads();

    // Mask + running top-8: 2 lanes per row, 8 columns each.
#pragma unroll
    for (int c = 0; c < 8; ++c) {
      const int cc = cbase + c;
      float s = stile[w][myrow][cc];
      if (s >= 0.0f && s <= 0.999f)          // threshold + exclude_self
        topk_insert(ts, ti, s, nb + cc);
    }
  }

  // Merge the two per-row partial lists (lane r and lane r+16) via LDS.
  if (hsel) {
#pragma unroll
    for (int j = 0; j < TOPK; ++j) { msim[w][myrow][j] = ts[j]; midx[w][myrow][j] = ti[j]; }
  }
  __syncthreads();
  if (!hsel) {
#pragma unroll
    for (int j = 0; j < TOPK; ++j)
      topk_insert(ts, ti, msim[w][myrow][j], midx[w][myrow][j]);
    const size_t base = ((size_t)(row0 + myrow) * NCHUNKS + chunk) * TOPK;
#pragma unroll
    for (int j = 0; j < TOPK; ++j) { psims[base + j] = ts[j]; pidx[base + j] = ti[j]; }
  }
}

// ---------------------------------------------------------------------------
// Kernel 3: merge per-chunk candidates, emit top_sims/mask, gather rows
// ---------------------------------------------------------------------------
__launch_bounds__(128)
__global__ void finalize_kernel(const float* __restrict__ psims,
                                const int* __restrict__ pidx,
                                const float* __restrict__ mem,
                                float* __restrict__ out_ret,
                                float* __restrict__ out_sims,
                                float* __restrict__ out_mask) {
  const int row = blockIdx.x;
  __shared__ float fs[TOPK];
  __shared__ int   fi[TOPK];
  if (threadIdx.x == 0) {
    float ts[TOPK]; int ti[TOPK];
#pragma unroll
    for (int j = 0; j < TOPK; ++j) { ts[j] = -INFINITY; ti[j] = 0; }
    const size_t base = (size_t)row * NCHUNKS * TOPK;
    for (int c = 0; c < NCHUNKS * TOPK; ++c)
      topk_insert(ts, ti, psims[base + c], pidx[base + c]);
#pragma unroll
    for (int j = 0; j < TOPK; ++j) { fs[j] = ts[j]; fi[j] = ti[j]; }
  }
  __syncthreads();
  if (threadIdx.x < TOPK) {
    float s = fs[threadIdx.x];
    out_sims[(size_t)row * TOPK + threadIdx.x] = s;
    out_mask[(size_t)row * TOPK + threadIdx.x] = (s > -INFINITY) ? 1.0f : 0.0f;
  }
  for (int i = threadIdx.x; i < TOPK * DIM; i += 128) {
    const int k = i >> 9;           // /DIM
    const int d = i & (DIM - 1);
    out_ret[(size_t)row * TOPK * DIM + i] = mem[(size_t)fi[k] * DIM + d];
  }
}

// ---------------------------------------------------------------------------
extern "C" void kernel_launch(void* const* d_in, const int* in_sizes, int n_in,
                              void* d_out, int out_size, void* d_ws, size_t ws_size,
                              hipStream_t stream) {
  (void)in_sizes; (void)n_in; (void)out_size;

  const float* q   = (const float*)d_in[0];   // [2048, 512]
  const float* mem = (const float*)d_in[1];   // [65536, 512] (unit-norm rows)

  float* out       = (float*)d_out;
  float* out_ret   = out;                                   // [2048, 8, 512]
  float* out_sims  = out + (size_t)B_Q * TOPK * DIM;        // [2048, 8]
  float* out_mask  = out_sims + (size_t)B_Q * TOPK;         // [2048, 8]

  const size_t bankB  = (size_t)N_MEM * DIM * sizeof(__bf16);  // 64 MB
  const size_t qB     = (size_t)B_Q * DIM * sizeof(__bf16);    //  2 MB
  const size_t qnB    = (size_t)B_Q * DIM * sizeof(float);     //  4 MB
  const size_t partB  = (size_t)B_Q * NCHUNKS * TOPK * sizeof(float); // 1 MB
  const size_t needFast = 2 * bankB + 2 * qB + qnB + 2 * partB;       // ~138 MB

  char* ws = (char*)d_ws;

  if (ws_size >= needFast) {
    // ---- fast path: pre-split bank+queries into bf16 hi/lo planes ----
    __bf16* bhi   = (__bf16*)ws;
    __bf16* blo   = (__bf16*)(ws + bankB);
    __bf16* qhi   = (__bf16*)(ws + 2 * bankB);
    __bf16* qlo   = (__bf16*)(ws + 2 * bankB + qB);
    float*  qn    = (float*)(ws + 2 * bankB + 2 * qB);
    float*  psims = (float*)(ws + 2 * bankB + 2 * qB + qnB);
    int*    pidx  = (int*)(ws + 2 * bankB + 2 * qB + qnB + partB);

    convert_bank_kernel<<<(N_MEM * (size_t)DIM) / (256 * 4), 256, 0, stream>>>(mem, bhi, blo);
    normalize_q_kernel<<<B_Q, 128, 0, stream>>>(q, qn, qhi, qlo);

    dim3 g2(NCHUNKS, MGROUPS);
    gemm_topk_kernel<true><<<g2, WAVES_PER_WG * 32, 0, stream>>>(
        qn, qhi, qlo, mem, bhi, blo, psims, pidx);
    finalize_kernel<<<B_Q, 128, 0, stream>>>(psims, pidx, mem,
                                             out_ret, out_sims, out_mask);
  } else {
    // ---- fallback: split in-flight inside the GEMM (small workspace) ----
    float* qn    = (float*)ws;
    float* psims = (float*)(ws + qnB);
    int*   pidx  = (int*)(ws + qnB + partB);

    normalize_q_kernel<<<B_Q, 128, 0, stream>>>(q, qn, nullptr, nullptr);

    dim3 g2(NCHUNKS, MGROUPS);
    gemm_topk_kernel<false><<<g2, WAVES_PER_WG * 32, 0, stream>>>(
        qn, nullptr, nullptr, mem, nullptr, nullptr, psims, pidx);
    finalize_kernel<<<B_Q, 128, 0, stream>>>(psims, pidx, mem,
                                             out_ret, out_sims, out_mask);
  }
}
